// SingleStreamMultiAttention_4174708211753
// MI455X (gfx1250) — matrix operand order repaired and checked by reading, not compiled, not run on hardware
//
#include <hip/hip_runtime.h>
#include <stdint.h>

typedef unsigned short u16;
typedef unsigned int   u32;
typedef unsigned long long u64;
typedef __attribute__((ext_vector_type(4)))  u32    v4u;
typedef __attribute__((ext_vector_type(16))) __bf16 v16bf;
typedef __attribute__((ext_vector_type(8)))  float  v8f;

union Frag { v4u u[2]; v16bf v; };

__device__ __forceinline__ u16 f2bf(float f) {
  u32 u = __float_as_uint(f);
  u32 r = u + 0x7FFFu + ((u >> 16) & 1u);
  return (u16)(r >> 16);
}

// LDS byte offset of a shared-memory pointer (aperture is in high 32 bits).
__device__ __forceinline__ u32 lds_off(const void* p) {
  return (u32)(uintptr_t)p;
}

// async global->LDS, 16 bytes per lane, tracked by ASYNCcnt
__device__ __forceinline__ void async_b128(u32 lds_addr, const void* gaddr) {
  asm volatile("global_load_async_to_lds_b128 %0, %1, off"
               :: "v"(lds_addr), "v"((u64)gaddr) : "memory");
}
__device__ __forceinline__ void wait_async0() {
  asm volatile("s_wait_asynccnt 0" ::: "memory");
}

// ------------------------- f32 -> bf16 convert -------------------------
__global__ __launch_bounds__(256) void k_conv(const float* __restrict__ in,
                                              u16* __restrict__ out, int n) {
  int i = blockIdx.x * 256 + threadIdx.x;
  if (i < n) out[i] = f2bf(in[i]);
}

// ------------------------- min/max over attn map -----------------------
__global__ __launch_bounds__(256) void k_minmax(const float* __restrict__ am,
                                                float* __restrict__ mm) {
  __shared__ float r0[256], r1[256], r2[256], r3[256];
  int tid = threadIdx.x;
  float mn0 = 3.4e38f, mx0 = -3.4e38f, mn1 = 3.4e38f, mx1 = -3.4e38f;
  for (int i = tid; i < 8192; i += 256) {
    float a = am[i], b = am[8192 + i];
    mn0 = fminf(mn0, a); mx0 = fmaxf(mx0, a);
    mn1 = fminf(mn1, b); mx1 = fmaxf(mx1, b);
  }
  r0[tid] = mn0; r1[tid] = mx0; r2[tid] = mn1; r3[tid] = mx1;
  __syncthreads();
  for (int s = 128; s > 0; s >>= 1) {
    if (tid < s) {
      r0[tid] = fminf(r0[tid], r0[tid + s]);
      r1[tid] = fmaxf(r1[tid], r1[tid + s]);
      r2[tid] = fminf(r2[tid], r2[tid + s]);
      r3[tid] = fmaxf(r3[tid], r3[tid + s]);
    }
    __syncthreads();
  }
  if (tid == 0) { mm[0] = r0[0]; mm[1] = r1[0]; mm[2] = r2[0]; mm[3] = r3[0]; }
}

// ------------------------- RoPE positions ------------------------------
__global__ __launch_bounds__(256) void k_pos(const float* __restrict__ am,
                                             const float* __restrict__ mm,
                                             float* __restrict__ pos) {
  int t = blockIdx.x * 256 + threadIdx.x;
  float a0 = am[t], a1 = am[8192 + t];
  float h1 = (a0 - mm[0]) / (mm[1] - mm[0] + 1e-8f) * 4.0f;
  float h2 = (a1 - mm[2]) / (mm[3] - mm[2] + 1e-8f) * 4.0f + 20.0f;
  pos[t] = (a0 >= a1) ? h1 : h2;   // argmax over 2 rows, ties -> row 0
}

// ------------------------- bf16 GEMM: C = A @ W^T + bias ---------------
// A: (M,K) bf16 row-major. W: (N,K) bf16 row-major. C: (M,N) f32.
// Block tile 128x128, 8 waves x (2x4) 16x16 WMMA tiles, K-step 32.
// Double-buffered LDS filled with GLOBAL_LOAD_ASYNC_TO_LDS_B128;
// one barrier per K-step, prefetch overlaps the WMMAs.
__global__ __launch_bounds__(256) void k_gemm(const u16* __restrict__ A,
                                              const u16* __restrict__ W,
                                              const float* __restrict__ bias,
                                              float* __restrict__ C,
                                              int M, int N, int K) {
  __shared__ __align__(16) u16 lA[2][128 * 40];
  __shared__ __align__(16) u16 lB[2][128 * 40];
  const int tid = threadIdx.x;
  const int bm = blockIdx.y * 128;
  const int bn = blockIdx.x * 128;
  const int wave = tid >> 5;
  const int lane = tid & 31;
  const int l16 = lane & 15;
  const int hi = lane >> 4;
  const int ah = hi * 8;     // A-fragment K half offset
  const int bh = hi * 16;    // unused (B fragments use ah too)
  const int wm = (wave >> 1) * 32;
  const int wn = (wave & 1) * 64;

  // per-thread staging coordinates (two 16B chunks per tile per thread)
  const int r0 = tid >> 2,          c0 = (tid & 3) * 8;
  const int r1 = (256 + tid) >> 2,  c1 = (tid & 3) * 8;
  const u16* gA0 = &A[(size_t)(bm + r0) * K + c0];
  const u16* gA1 = &A[(size_t)(bm + r1) * K + c1];
  const u16* gB0 = &W[(size_t)(bn + r0) * K + c0];
  const u16* gB1 = &W[(size_t)(bn + r1) * K + c1];
  u32 dA0[2], dA1[2], dB0[2], dB1[2];
#pragma unroll
  for (int b = 0; b < 2; ++b) {
    dA0[b] = lds_off(&lA[b][r0 * 40 + c0]);
    dA1[b] = lds_off(&lA[b][r1 * 40 + c1]);
    dB0[b] = lds_off(&lB[b][r0 * 40 + c0]);
    dB1[b] = lds_off(&lB[b][r1 * 40 + c1]);
  }

  v8f zero = {0.f, 0.f, 0.f, 0.f, 0.f, 0.f, 0.f, 0.f};
  v8f acc[2][4];
#pragma unroll
  for (int i = 0; i < 2; ++i)
#pragma unroll
    for (int j = 0; j < 4; ++j) acc[i][j] = zero;

  const int steps = K >> 5;
  // preload buffer 0
  async_b128(dA0[0], gA0);
  async_b128(dA1[0], gA1);
  async_b128(dB0[0], gB0);
  async_b128(dB1[0], gB1);

  for (int it = 0; it < steps; ++it) {
    const int buf = it & 1;
    wait_async0();        // this wave's async writes to `buf` landed
    __syncthreads();      // everyone's landed; last iter's ds reads done
    if (it + 1 < steps) { // prefetch next K-step into the other buffer
      const int k0 = (it + 1) * 32;
      async_b128(dA0[buf ^ 1], gA0 + k0);
      async_b128(dA1[buf ^ 1], gA1 + k0);
      async_b128(dB0[buf ^ 1], gB0 + k0);
      async_b128(dB1[buf ^ 1], gB1 + k0);
    }
    Frag fa[2], fb[4];
#pragma unroll
    for (int i = 0; i < 2; ++i) {
      int r = (wm + i * 16 + l16) * 40;
      fa[i].u[0] = *(const v4u*)&lA[buf][r + ah];
      fa[i].u[1] = *(const v4u*)&lA[buf][r + 16 + ah];
    }
#pragma unroll
    for (int j = 0; j < 4; ++j) {
      int r = (wn + j * 16 + l16) * 40;
      fb[j].u[0] = *(const v4u*)&lB[buf][r + ah];
      fb[j].u[1] = *(const v4u*)&lB[buf][r + 16 + ah];
    }
#pragma unroll
    for (int i = 0; i < 2; ++i)
#pragma unroll
      for (int j = 0; j < 4; ++j)
        acc[i][j] = __builtin_amdgcn_wmma_f32_16x16x32_bf16(
            false, fa[i].v, false, fb[j].v, (short)0, acc[i][j], false, false);
  }

#pragma unroll
  for (int j = 0; j < 4; ++j) {
    int n = bn + wn + j * 16 + l16;
    float bv = bias[n];
#pragma unroll
    for (int i = 0; i < 2; ++i) {
#pragma unroll
      for (int v = 0; v < 8; ++v) {
        int m = bm + wm + i * 16 + v + hi * 8;
        C[(size_t)m * N + n] = acc[i][j][v] + bv;
      }
    }
  }
}

// ------------------------- RoPE on q (f32 -> bf16) ---------------------
__global__ __launch_bounds__(256) void k_rope_q(const float* __restrict__ q,
                                                const float* __restrict__ pos,
                                                u16* __restrict__ qbf) {
  int idx = blockIdx.x * 256 + threadIdx.x;   // 8192*1024 pairs
  int t = idx >> 10, p = idx & 1023;
  int c = p * 2;
  int i = (c & 127) >> 1;                     // freq index in head
  float freq = __expf(-(float)i * (9.210340371976184f / 64.0f)); // 10000^{-i/64}
  float ang = pos[t] * freq;
  float sn, cs;
  __sincosf(ang, &sn, &cs);
  size_t o = (size_t)t * 2048 + c;
  float e = q[o], od = q[o + 1];
  qbf[o] = f2bf(e * cs - od * sn);
  qbf[o + 1] = f2bf(od * cs + e * sn);
}

// ------------------------- RoPE on k (kv f32 -> k bf16) ----------------
__global__ __launch_bounds__(256) void k_rope_k(const float* __restrict__ kv,
                                                u16* __restrict__ kbf) {
  int idx = blockIdx.x * 256 + threadIdx.x;   // 256*1024 pairs
  int t = idx >> 10, p = idx & 1023;
  int c = p * 2;
  int i = (c & 127) >> 1;
  float freq = __expf(-(float)i * (9.210340371976184f / 64.0f));
  float pz = ((t & 31) < 16) ? 2.0f : 22.0f;  // encoder positions
  float ang = pz * freq;
  float sn, cs;
  __sincosf(ang, &sn, &cs);
  size_t src = (size_t)t * 4096 + c;
  float e = kv[src], od = kv[src + 1];
  size_t dst = (size_t)t * 2048 + c;
  kbf[dst] = f2bf(e * cs - od * sn);
  kbf[dst + 1] = f2bf(od * cs + e * sn);
}

// ------------------------- v slice convert -----------------------------
__global__ __launch_bounds__(256) void k_conv_v(const float* __restrict__ kv,
                                                u16* __restrict__ vbf) {
  int idx = blockIdx.x * 256 + threadIdx.x;   // 256*2048
  int t = idx >> 11, c = idx & 2047;
  vbf[idx] = f2bf(kv[(size_t)t * 4096 + 2048 + c]);
}

// ------------------------- attention (WMMA) ----------------------------
// Block = (frame f, head h). q:(8192,2048)bf16, k/v:(256,2048)bf16.
// Per wave: 128 query rows as 8 m-tiles; S = QK^T via WMMA, half-wave
// shuffle softmax, P transposed through per-wave LDS tile, O = P V via WMMA.
__global__ __launch_bounds__(256) void k_attn(const u16* __restrict__ qbf,
                                              const u16* __restrict__ kbf,
                                              const u16* __restrict__ vbf,
                                              u16* __restrict__ obf) {
  __shared__ __align__(16) u16 lK[32 * 136];     // 32 keys x 128 d (pitch 136)
  __shared__ __align__(16) u16 lVT[128 * 40];    // 128 d x 32 keys (pitch 40)
  __shared__ __align__(16) u16 lP[8][16 * 40];   // per-wave P staging
  const int f = blockIdx.x >> 4;
  const int h = blockIdx.x & 15;
  const int tid = threadIdx.x;
  const int wave = tid >> 5;
  const int lane = tid & 31;
  const int l16 = lane & 15;
  const int hi = lane >> 4;
  const int ah = hi * 8;
  const int bh = hi * 16;

#pragma unroll
  for (int c = 0; c < 2; ++c) {          // stage K (row-major keys x d)
    int idx = c * 256 + tid;
    int row = idx >> 4;
    int col = (idx & 15) * 8;
    *(v4u*)&lK[row * 136 + col] =
        *(const v4u*)&kbf[(size_t)(f * 32 + row) * 2048 + h * 128 + col];
  }
#pragma unroll
  for (int c = 0; c < 16; ++c) {         // stage V transposed (d x keys)
    int idx = c * 256 + tid;
    int a = idx >> 7, d = idx & 127;
    lVT[d * 40 + a] = vbf[(size_t)(f * 32 + a) * 2048 + h * 128 + d];
  }
  __syncthreads();

  const float scale = 0.08838834764831845f;  // 1/sqrt(128)
  const v8f zero = {0.f, 0.f, 0.f, 0.f, 0.f, 0.f, 0.f, 0.f};
  u16* pw = &lP[wave][0];

  for (int mt = 0; mt < 8; ++mt) {
    int rowg = f * 1024 + wave * 128 + mt * 16 + l16;
    const u16* qrow = &qbf[(size_t)rowg * 2048 + h * 128];
    v8f s0 = zero, s1 = zero;
#pragma unroll
    for (int kt = 0; kt < 4; ++kt) {
      Frag fa, fb0, fb1;
      fa.u[0] = *(const v4u*)&qrow[kt * 32 + ah];
      fa.u[1] = *(const v4u*)&qrow[kt * 32 + 16 + ah];
      int rb0 = l16 * 136 + kt * 32 + bh;
      int rb1 = (16 + l16) * 136 + kt * 32 + bh;
      fb0.u[0] = *(const v4u*)&lK[rb0 - bh + ah];
      fb0.u[1] = *(const v4u*)&lK[rb0 - bh + 16 + ah];
      fb1.u[0] = *(const v4u*)&lK[rb1 - bh + ah];
      fb1.u[1] = *(const v4u*)&lK[rb1 - bh + 16 + ah];
      s0 = __builtin_amdgcn_wmma_f32_16x16x32_bf16(false, fa.v, false, fb0.v,
                                                   (short)0, s0, false, false);
      s1 = __builtin_amdgcn_wmma_f32_16x16x32_bf16(false, fa.v, false, fb1.v,
                                                   (short)0, s1, false, false);
    }
    // softmax over 32 keys; each score row lives in one 16-lane half
    float p0[8], p1[8];
#pragma unroll
    for (int v = 0; v < 8; ++v) {
      float x0 = s0[v] * scale, x1 = s1[v] * scale;
      float mx = fmaxf(x0, x1);
#pragma unroll
      for (int o = 1; o < 16; o <<= 1) mx = fmaxf(mx, __shfl_xor(mx, o, 32));
      float e0 = __expf(x0 - mx), e1 = __expf(x1 - mx);
      float sm = e0 + e1;
#pragma unroll
      for (int o = 1; o < 16; o <<= 1) sm += __shfl_xor(sm, o, 32);
      float inv = 1.0f / sm;
      p0[v] = e0 * inv;
      p1[v] = e1 * inv;
    }
    // transpose P (C layout -> A layout) through per-wave LDS tile;
    // same-wave LDS ops are in-order so no barrier needed.
#pragma unroll
    for (int v = 0; v < 8; ++v) {
      int m = v + hi * 8;
      pw[m * 40 + l16] = f2bf(p0[v]);
      pw[m * 40 + 16 + l16] = f2bf(p1[v]);
    }
    Frag fp;
    fp.u[0] = *(const v4u*)&pw[l16 * 40 + ah];
    fp.u[1] = *(const v4u*)&pw[l16 * 40 + 16 + ah];
#pragma unroll
    for (int nt = 0; nt < 8; ++nt) {
      Frag fv;
      int r = (nt * 16 + l16) * 40;
      fv.u[0] = *(const v4u*)&lVT[r + ah];
      fv.u[1] = *(const v4u*)&lVT[r + 16 + ah];
      v8f oacc = __builtin_amdgcn_wmma_f32_16x16x32_bf16(
          false, fp.v, false, fv.v, (short)0, zero, false, false);
#pragma unroll
      for (int v = 0; v < 8; ++v) {
        int m = wave * 128 + mt * 16 + v + hi * 8;
        obf[(size_t)(f * 1024 + m) * 2048 + h * 128 + nt * 16 + l16] =
            f2bf(oacc[v]);
      }
    }
  }
}

// ------------------------- host orchestration --------------------------
extern "C" void kernel_launch(void* const* d_in, const int* in_sizes, int n_in,
                              void* d_out, int out_size, void* d_ws, size_t ws_size,
                              hipStream_t stream) {
  (void)in_sizes; (void)n_in; (void)out_size; (void)ws_size;
  const float* x   = (const float*)d_in[0];
  const float* enc = (const float*)d_in[1];
  const float* am  = (const float*)d_in[2];
  const float* Wq  = (const float*)d_in[3];
  const float* bq  = (const float*)d_in[4];
  const float* Wkv = (const float*)d_in[5];
  const float* bkv = (const float*)d_in[6];
  const float* Wp  = (const float*)d_in[7];
  const float* bp  = (const float*)d_in[8];
  float* out = (float*)d_out;

  char* w = (char*)d_ws;
  u16*   x_bf   = (u16*)w;   w += (size_t)16777216 * 2;   // x bf16 (8192x2048)
  float* q_f    = (float*)w; w += (size_t)16777216 * 4;   // q f32
  u16*   Wq_bf  = (u16*)w;   w += (size_t)4194304 * 2;
  u16*   Wkv_bf = (u16*)w;   w += (size_t)3145728 * 2;
  u16*   Wp_bf  = (u16*)w;   w += (size_t)4194304 * 2;
  u16*   enc_bf = (u16*)w;   w += (size_t)196608 * 2;
  float* kv_f   = (float*)w; w += (size_t)1048576 * 4;    // 256x4096
  u16*   k_bf   = (u16*)w;   w += (size_t)524288 * 2;     // 256x2048
  u16*   v_bf   = (u16*)w;   w += (size_t)524288 * 2;
  float* pos    = (float*)w; w += (size_t)8192 * 4;
  float* mm     = (float*)w; w += 256;
  u16* q_bf = x_bf;          // alias: x_bf dead after Q GEMM
  u16* o_bf = (u16*)q_f;     // alias: q_f dead after rope_q

  k_conv<<<65536, 256, 0, stream>>>(x, x_bf, 16777216);
  k_conv<<<16384, 256, 0, stream>>>(Wq, Wq_bf, 4194304);
  k_conv<<<12288, 256, 0, stream>>>(Wkv, Wkv_bf, 3145728);
  k_conv<<<16384, 256, 0, stream>>>(Wp, Wp_bf, 4194304);
  k_conv<<<768, 256, 0, stream>>>(enc, enc_bf, 196608);
  k_minmax<<<1, 256, 0, stream>>>(am, mm);
  k_pos<<<32, 256, 0, stream>>>(am, mm, pos);
  // q = x @ Wq^T + bq
  k_gemm<<<dim3(16, 64), 256, 0, stream>>>(x_bf, Wq_bf, bq, q_f, 8192, 2048, 2048);
  k_rope_q<<<32768, 256, 0, stream>>>(q_f, pos, q_bf);
  // kv = enc @ Wkv^T + bkv
  k_gemm<<<dim3(32, 2), 256, 0, stream>>>(enc_bf, Wkv_bf, bkv, kv_f, 256, 4096, 768);
  k_rope_k<<<1024, 256, 0, stream>>>(kv_f, k_bf);
  k_conv_v<<<2048, 256, 0, stream>>>(kv_f, v_bf);
  k_attn<<<128, 256, 0, stream>>>(q_bf, k_bf, v_bf, o_bf);
  // out = o @ Wp^T + bp
  k_gemm<<<dim3(16, 64), 256, 0, stream>>>(o_bf, Wp_bf, bp, out, 8192, 2048, 2048);
}
